// Model_72438918414716
// MI455X (gfx1250) — matrix operand order repaired and checked by reading, not compiled, not run on hardware
//
#include <hip/hip_runtime.h>
#include <hip/hip_bf16.h>

// ---------------------------------------------------------------------------
// 2-layer RNN (H=4096, batch=3, T=25) + per-step FC head for MI455X (gfx1250).
// Memory-bound on weight streaming -> convert weights to bf16 once per call
// (100MB, fits the 192MB L2), run the recurrent GEMVs with
// v_wmma_f32_16x16x32_bf16 (fp32 accumulate), keep h / biases / tanh / FC fp32.
// Hidden state is padded to 16 batch rows (rows 3..15 zero) so every lane's
// B-fragment load is unconditional -> no per-iteration EXEC save/restore.
// ---------------------------------------------------------------------------

#define H 4096
#define T 25
#define B 3
#define BPAD 16          // padded batch rows for the WMMA B operand
#define NOUT 19

typedef __attribute__((ext_vector_type(16))) __bf16 v16bf;
typedef __attribute__((ext_vector_type(8)))  float  v8f;

struct BF16Frag { uint4 lo, hi; };   // 32 bytes = 16 bf16 per lane

__device__ __forceinline__ unsigned short f2bf(float f) {
    unsigned int u = __float_as_uint(f);
    unsigned int r = 0x7FFFu + ((u >> 16) & 1u);   // round-to-nearest-even
    return (unsigned short)((u + r) >> 16);
}

// A fragment: 16x32 bf16, row-major W, stride H. Per ISA 16-bit A layout:
// lane L (half = L>>4, m = L&15) holds K = [kb + half*8, +8) in v[0:4]
// and K = [kb + 16 + half*8, +8) in v[4:8]  -> two contiguous 16B loads.
__device__ __forceinline__ v16bf load_a_frag(const unsigned short* __restrict__ wrow,
                                             int kb, int half) {
    BF16Frag f;
    f.lo = *(const uint4*)(wrow + kb + half * 8);
    f.hi = *(const uint4*)(wrow + kb + 16 + half * 8);
    return __builtin_bit_cast(v16bf, f);
}

// B fragment: 32x16 bf16 (K x N), N = batch column. Per ISA 16-bit B layout:
// lane L (n = L&15, half = L>>4) holds K = [kb + half*8, +8) and
// K = [kb + 16 + half*8, +8) of column n (matches A). hpad has 16 rows; rows >= 3 are
// zeroed once per call, so the load needs no predication.
__device__ __forceinline__ v16bf load_b_frag(const unsigned short* __restrict__ hpad,
                                             int nb, int kb, int half) {
    const unsigned short* p = hpad + nb * H + kb + half * 8;
    BF16Frag f;
    f.lo = *(const uint4*)(p);
    f.hi = *(const uint4*)(p + 16);
    return __builtin_bit_cast(v16bf, f);
}

// ---------------------------------------------------------------------------
// fp32 -> bf16 weight conversion, 8 elements / thread
// ---------------------------------------------------------------------------
__global__ __launch_bounds__(256) void cvt_w_bf16(const float* __restrict__ src,
                                                  unsigned short* __restrict__ dst,
                                                  int n8) {
    int i = blockIdx.x * blockDim.x + threadIdx.x;
    if (i >= n8) return;
    int base = i * 8;
    float4 a = *(const float4*)(src + base);
    float4 b = *(const float4*)(src + base + 4);
    uint4 o;
    o.x = (unsigned)f2bf(a.x) | ((unsigned)f2bf(a.y) << 16);
    o.y = (unsigned)f2bf(a.z) | ((unsigned)f2bf(a.w) << 16);
    o.z = (unsigned)f2bf(b.x) | ((unsigned)f2bf(b.y) << 16);
    o.w = (unsigned)f2bf(b.z) | ((unsigned)f2bf(b.w) << 16);
    *(uint4*)(dst + base) = o;
}

// Zero both double-buffered padded hidden arrays (h re-inits to zero every
// forward; pad rows 3..15 are never written again and stay zero all 25 steps).
__global__ __launch_bounds__(256) void init_h_zero(uint4* __restrict__ h0,
                                                   uint4* __restrict__ h1,
                                                   int n16) {
    int i = blockIdx.x * blockDim.x + threadIdx.x;
    if (i < n16) {
        uint4 z = make_uint4(0u, 0u, 0u, 0u);
        h0[i] = z;
        h1[i] = z;
    }
}

// ---------------------------------------------------------------------------
// Layer 0: h0n = tanh(x_t * W_ih0 + b_ih0 + h0_prev @ W_hh0^T + b_hh0)
// One wave per 16-row output tile; K-loop of 128 WMMAs.
// ---------------------------------------------------------------------------
__global__ __launch_bounds__(256) void rnn_layer0_kernel(
    const unsigned short* __restrict__ Whh,   // [H][H] bf16
    const float* __restrict__ Wih0,           // [H]
    const float* __restrict__ bih, const float* __restrict__ bhh,
    const float* __restrict__ x,              // [B][1][T]
    const unsigned short* __restrict__ hprev, // [BPAD][H] bf16
    unsigned short* __restrict__ hnext,       // [BPAD][H] bf16
    float* __restrict__ hf32,                 // [B][H] fp32 (d_out hidden, layer 0)
    int t)
{
    int wave = threadIdx.x >> 5;
    int lane = threadIdx.x & 31;
    int tile = blockIdx.x * (blockDim.x >> 5) + wave;   // 0..255
    int nb   = lane & 15;
    int half = lane >> 4;

    const unsigned short* wrow = Whh + (size_t)(tile * 16 + nb) * H;

    v8f c = {};
    for (int kb = 0; kb < H; kb += 32) {
        v16bf a = load_a_frag(wrow, kb, half);
        v16bf b = load_b_frag(hprev, nb, kb, half);
        c = __builtin_amdgcn_wmma_f32_16x16x32_bf16(false, a, false, b,
                                                    (short)0, c, false, false);
    }

    if (nb < B) {
        float xv = x[nb * T + t];               // x[b, 0, t]
        int mbase = tile * 16 + half * 8;
        #pragma unroll
        for (int r = 0; r < 8; ++r) {
            int m = mbase + r;
            float h = tanhf(c[r] + xv * Wih0[m] + bih[m] + bhh[m]);
            hnext[nb * H + m] = f2bf(h);
            hf32[nb * H + m]  = h;
        }
    }
}

// ---------------------------------------------------------------------------
// Layer 1: h1n = tanh(h0n @ W_ih1^T + b_ih1 + h1_prev @ W_hh1^T + b_hh1)
// Two chained WMMAs per k-step into one fp32 accumulator.
// ---------------------------------------------------------------------------
__global__ __launch_bounds__(256) void rnn_layer1_kernel(
    const unsigned short* __restrict__ Wih,   // [H][H] bf16
    const unsigned short* __restrict__ Whh,   // [H][H] bf16
    const float* __restrict__ bih, const float* __restrict__ bhh,
    const unsigned short* __restrict__ h0new, // [BPAD][H] bf16 (layer-0 out, this step)
    const unsigned short* __restrict__ h1prev,// [BPAD][H] bf16
    unsigned short* __restrict__ h1next,      // [BPAD][H] bf16
    float* __restrict__ hf32)                 // [B][H] fp32 (d_out hidden, layer 1)
{
    int wave = threadIdx.x >> 5;
    int lane = threadIdx.x & 31;
    int tile = blockIdx.x * (blockDim.x >> 5) + wave;
    int nb   = lane & 15;
    int half = lane >> 4;

    const unsigned short* wirow = Wih + (size_t)(tile * 16 + nb) * H;
    const unsigned short* whrow = Whh + (size_t)(tile * 16 + nb) * H;

    v8f c = {};
    for (int kb = 0; kb < H; kb += 32) {
        v16bf ai = load_a_frag(wirow, kb, half);
        v16bf bi = load_b_frag(h0new, nb, kb, half);
        c = __builtin_amdgcn_wmma_f32_16x16x32_bf16(false, ai, false, bi,
                                                    (short)0, c, false, false);
        v16bf ah = load_a_frag(whrow, kb, half);
        v16bf bh = load_b_frag(h1prev, nb, kb, half);
        c = __builtin_amdgcn_wmma_f32_16x16x32_bf16(false, ah, false, bh,
                                                    (short)0, c, false, false);
    }

    if (nb < B) {
        int mbase = tile * 16 + half * 8;
        #pragma unroll
        for (int r = 0; r < 8; ++r) {
            int m = mbase + r;
            float h = tanhf(c[r] + bih[m] + bhh[m]);
            h1next[nb * H + m] = f2bf(h);
            hf32[nb * H + m]   = h;
        }
    }
}

// ---------------------------------------------------------------------------
// FC head: out[b, t, o] = dot(fc_W[o], h1[b]) + fc_b[o]   (57 blocks / step)
// ---------------------------------------------------------------------------
__global__ __launch_bounds__(256) void fc_head_kernel(
    const float* __restrict__ fcW, const float* __restrict__ fcb,
    const float* __restrict__ h1, float* __restrict__ outs, int t)
{
    __shared__ float red[256];
    int o = blockIdx.x % NOUT;
    int b = blockIdx.x / NOUT;
    const float* w = fcW + o * H;
    const float* h = h1 + b * H;
    float s = 0.f;
    for (int k = threadIdx.x; k < H; k += 256) s += w[k] * h[k];
    red[threadIdx.x] = s;
    __syncthreads();
    for (int off = 128; off > 0; off >>= 1) {
        if ((int)threadIdx.x < off) red[threadIdx.x] += red[threadIdx.x + off];
        __syncthreads();
    }
    if (threadIdx.x == 0) outs[(b * T + t) * NOUT + o] = red[0] + fcb[o];
}

// ---------------------------------------------------------------------------
// Launch
// ---------------------------------------------------------------------------
extern "C" void kernel_launch(void* const* d_in, const int* in_sizes, int n_in,
                              void* d_out, int out_size, void* d_ws, size_t ws_size,
                              hipStream_t stream) {
    (void)in_sizes; (void)n_in; (void)out_size; (void)ws_size;

    const float* x     = (const float*)d_in[0];   // [3,1,25]
    const float* W_ih0 = (const float*)d_in[1];   // [H,1]
    const float* W_hh0 = (const float*)d_in[2];   // [H,H]
    const float* b_ih0 = (const float*)d_in[3];
    const float* b_hh0 = (const float*)d_in[4];
    const float* W_ih1 = (const float*)d_in[5];   // [H,H]
    const float* W_hh1 = (const float*)d_in[6];   // [H,H]
    const float* b_ih1 = (const float*)d_in[7];
    const float* b_hh1 = (const float*)d_in[8];
    const float* fc_W  = (const float*)d_in[9];   // [19,H]
    const float* fc_b  = (const float*)d_in[10];

    // workspace layout
    const size_t WELEMS = (size_t)H * H;          // 16M elements per matrix
    const size_t HBUF   = (size_t)BPAD * H;       // one padded hidden buffer (elems)
    char* ws = (char*)d_ws;
    unsigned short* whh0_bf = (unsigned short*)(ws);
    unsigned short* wih1_bf = (unsigned short*)(ws + WELEMS * 2);
    unsigned short* whh1_bf = (unsigned short*)(ws + 2 * WELEMS * 2);
    unsigned short* h0_bf   = (unsigned short*)(ws + 3 * WELEMS * 2);   // [2][BPAD][H]
    unsigned short* h1_bf   = h0_bf + 2 * HBUF;                         // [2][BPAD][H]

    // d_out: outs [3][25][19] (1425 floats) then hidden [2][3][H]
    float* outp   = (float*)d_out;
    float* hidden = outp + B * T * NOUT;
    float* h0_f32 = hidden;                // layer 0 final hidden
    float* h1_f32 = hidden + B * H;        // layer 1 final hidden

    // 1) weight conversion fp32 -> bf16 (every call; no cached state)
    {
        int n8 = (int)(WELEMS / 8);
        int grid = (n8 + 255) / 256;
        cvt_w_bf16<<<grid, 256, 0, stream>>>(W_hh0, whh0_bf, n8);
        cvt_w_bf16<<<grid, 256, 0, stream>>>(W_ih1, wih1_bf, n8);
        cvt_w_bf16<<<grid, 256, 0, stream>>>(W_hh1, whh1_bf, n8);
    }
    // 2) zero both padded hidden double-buffers (rows 3..15 stay zero forever)
    {
        int n16 = (int)(2 * HBUF / 8);     // uint4 = 8 ushorts
        init_h_zero<<<(n16 + 255) / 256, 256, 0, stream>>>(
            (uint4*)h0_bf, (uint4*)h1_bf, n16);
    }

    // 3) 25 recurrent steps; 256 tiles -> 32 blocks x 8 waves
    const int TILE_BLOCKS = (H / 16) / 8;  // 32
    for (int t = 0; t < T; ++t) {
        int cur = t & 1, nxt = cur ^ 1;
        unsigned short* h0p = h0_bf + cur * HBUF;
        unsigned short* h0n = h0_bf + nxt * HBUF;
        unsigned short* h1p = h1_bf + cur * HBUF;
        unsigned short* h1n = h1_bf + nxt * HBUF;

        rnn_layer0_kernel<<<TILE_BLOCKS, 256, 0, stream>>>(
            whh0_bf, W_ih0, b_ih0, b_hh0, x, h0p, h0n, h0_f32, t);

        rnn_layer1_kernel<<<TILE_BLOCKS, 256, 0, stream>>>(
            wih1_bf, whh1_bf, b_ih1, b_hh1, h0n, h1p, h1n, h1_f32);

        fc_head_kernel<<<B * NOUT, 256, 0, stream>>>(
            fc_W, fc_b, h1_f32, outp, t);
    }
}
